// AttenComm_62534723829932
// MI455X (gfx1250) — matrix operand order repaired and checked by reading, not compiled, not run on hardware
//
#include <hip/hip_runtime.h>
#include <hip/hip_bf16.h>

#define B_ 4
#define L_ 5
#define C_ 256
#define H_ 64
#define W_ 256
#define HW_ (H_ * W_)
#define TILE 16
#define THREADS 256

typedef __attribute__((ext_vector_type(16))) __bf16 v16bf;
typedef __attribute__((ext_vector_type(8)))  float  v8f;

__device__ __forceinline__ unsigned short f2bf(float f) {
    unsigned u = __builtin_bit_cast(unsigned, f);
    u += 0x7FFFu + ((u >> 16) & 1u);          // round-to-nearest-even
    return (unsigned short)(u >> 16);
}
__device__ __forceinline__ float bf2f(unsigned short s) {
    unsigned u = ((unsigned)s) << 16;
    return __builtin_bit_cast(float, u);
}

// ---------------- rate reduction ----------------
__global__ void rate_partial_kernel(const float* __restrict__ rm, float* __restrict__ wsf) {
    __shared__ float red[256];
    int b = blockIdx.x;                               // batch
    const float* a0 = rm + (size_t)(b * L_ * 2) * HW_; // agent 0, anchor 0
    const float* a1 = a0 + HW_;                        // anchor 1
    int cnt = 0;
    for (int i = threadIdx.x; i < HW_; i += 256)
        cnt += (fmaxf(a0[i], a1[i]) > 0.0f) ? 1 : 0;   // sigmoid(z)>0.5 <=> z>0
    red[threadIdx.x] = (float)cnt;
    __syncthreads();
    for (int s = 128; s > 0; s >>= 1) {
        if (threadIdx.x < s) red[threadIdx.x] += red[threadIdx.x + s];
        __syncthreads();
    }
    if (threadIdx.x == 0) wsf[b] = red[0];
}

__global__ void rate_final_kernel(const float* __restrict__ wsf, float* __restrict__ out) {
    out[(size_t)B_ * C_ * HW_] =
        (wsf[0] + wsf[1] + wsf[2] + wsf[3]) * (1.0f / ((float)HW_ * (float)B_));
}

// ---------------- fused warp + per-pixel attention ----------------
__global__ void __launch_bounds__(THREADS)
fuse_kernel(const float* __restrict__ x, const float* __restrict__ rm,
            const float* __restrict__ T44, float* __restrict__ out)
{
    __shared__ unsigned short t_lds[TILE * L_ * C_];   // warped feats, bf16 bits, 40 KB
    __shared__ float w_lds[TILE][8];                   // softmax weights per pixel

    const int tilesPerRow = W_ / TILE;                 // 16
    int blk = blockIdx.x;
    int b   = blk / (H_ * tilesPerRow);
    int rem = blk % (H_ * tilesPerRow);
    int h   = rem / tilesPerRow;
    int w0  = (rem % tilesPerRow) * TILE;

    int tid = threadIdx.x;
    int p   = tid & (TILE - 1);                        // pixel within tile
    int cc  = tid >> 4;                                // channel group 0..15

    int w = w0 + p;
    float gx = -1.0f + 2.0f * (float)w / (float)(W_ - 1);
    float gy = -1.0f + 2.0f * (float)h / (float)(H_ - 1);

    bool oddb = (b & 1) != 0;
    // masks_flat[b] == batch0 agent b; even agent -> all ones
    const float* rm_m0 = rm + (size_t)(b * 2) * HW_;
    const float* rm_m1 = rm_m0 + HW_;

    // ---- phase 1: gather-warp all 5 agents for this 16-pixel tile into LDS ----
    for (int m = 0; m < L_; ++m) {
        const float* t44 = T44 + (size_t)(((b * L_ + 0) * L_ + m) * 16);
        float M00 = t44[0];
        float M01 = t44[1] * ((float)H_ / (float)W_);
        float M02 = t44[3] * (2.0f / (4.0f * 0.4f * (float)W_));
        float M10 = t44[4] * ((float)W_ / (float)H_);
        float M11 = t44[5];
        float M12 = t44[7] * (2.0f / (4.0f * 0.4f * (float)H_));

        float gxx = M00 * gx + M01 * gy + M02;
        float gyy = M10 * gx + M11 * gy + M12;
        float xs = (gxx + 1.0f) * 0.5f * (float)(W_ - 1);
        float ys = (gyy + 1.0f) * 0.5f * (float)(H_ - 1);
        float x0f = floorf(xs), y0f = floorf(ys);
        float wx1 = xs - x0f, wx0 = 1.0f - wx1;
        float wy1 = ys - y0f, wy0 = 1.0f - wy1;

        float tx[4]  = {x0f, x0f + 1.0f, x0f, x0f + 1.0f};
        float ty[4]  = {y0f, y0f, y0f + 1.0f, y0f + 1.0f};
        float twt[4] = {wx0 * wy0, wx1 * wy0, wx0 * wy1, wx1 * wy1};
        int   off[4];
#pragma unroll
        for (int t = 0; t < 4; ++t) {
            bool valid = (tx[t] >= 0.0f) && (tx[t] <= (float)(W_ - 1)) &&
                         (ty[t] >= 0.0f) && (ty[t] <= (float)(H_ - 1));
            int xi = min(max((int)tx[t], 0), W_ - 1);
            int yi = min(max((int)ty[t], 0), H_ - 1);
            int o  = yi * W_ + xi;
            float mk = 1.0f;
            if (oddb) mk = (fmaxf(rm_m0[o], rm_m1[o]) > 0.0f) ? 1.0f : 0.0f;
            twt[t] *= valid ? mk : 0.0f;
            off[t] = o;
        }
        const float* xb = x + (size_t)(b * L_ + m) * C_ * HW_;
        unsigned short* tb = &t_lds[(p * L_ + m) * C_];
#pragma unroll 4
        for (int j = 0; j < 16; ++j) {
            int c = cc * 16 + j;
            const float* xc = xb + (size_t)c * HW_;
            float v = twt[0] * xc[off[0]] + twt[1] * xc[off[1]] +
                      twt[2] * xc[off[2]] + twt[3] * xc[off[3]];
            tb[c] = f2bf(v);
        }
    }
    __syncthreads();

    // ---- phase 2: per-pixel 5x5 Gram via v_wmma_f32_16x16x32_bf16, softmax row 0 ----
    int wid  = tid >> 5;                               // wave 0..7, 2 pixels each
    int lane = tid & 31;
    int mrow = lane & 15;                              // A row / B col (agent index)
    int hi   = lane >> 4;

    for (int pp = 0; pp < 2; ++pp) {
        int px = wid * 2 + pp;
        v8f acc = {};
#pragma unroll
        for (int kb = 0; kb < C_; kb += 32) {
            union { unsigned short s[16]; v16bf v; } ua, ub;
            if (mrow < L_) {
                const unsigned short* ta = &t_lds[(px * L_ + mrow) * C_];
#pragma unroll
                for (int j = 0; j < 16; ++j) {         // A: K = hi*16+j
                    ua.s[j] = ta[kb + hi * 16 + j];
                }
#pragma unroll
                for (int j = 0; j < 16; ++j)           // B: K = hi*16+j (B = A^T)
                    ub.s[j] = ta[kb + hi * 16 + j];
            } else {
#pragma unroll
                for (int j = 0; j < 16; ++j) { ua.s[j] = 0; ub.s[j] = 0; }
            }
            acc = __builtin_amdgcn_wmma_f32_16x16x32_bf16(
                      false, ua.v, false, ub.v, (short)0, acc, false, false);
        }
        // score[0][n] sits in acc[0], lanes n=0..4; scale 1/sqrt(256)
        float s0 = __shfl(acc[0], 0, 32) * 0.0625f;
        float s1 = __shfl(acc[0], 1, 32) * 0.0625f;
        float s2 = __shfl(acc[0], 2, 32) * 0.0625f;
        float s3 = __shfl(acc[0], 3, 32) * 0.0625f;
        float s4 = __shfl(acc[0], 4, 32) * 0.0625f;
        float mx = fmaxf(fmaxf(fmaxf(s0, s1), fmaxf(s2, s3)), s4);
        float e0 = __expf(s0 - mx), e1 = __expf(s1 - mx), e2 = __expf(s2 - mx);
        float e3 = __expf(s3 - mx), e4 = __expf(s4 - mx);
        float inv = 1.0f / (e0 + e1 + e2 + e3 + e4);
        if (lane < 5) {
            float ev = (lane == 0) ? e0 : (lane == 1) ? e1 : (lane == 2) ? e2
                     : (lane == 3) ? e3 : e4;
            w_lds[px][lane] = ev * inv;
        }
    }
    __syncthreads();

    // ---- phase 3: weighted combine, coalesced store (half-wave = 16 consecutive w) ----
    float wgt[L_];
#pragma unroll
    for (int m = 0; m < L_; ++m) wgt[m] = w_lds[p][m];
    const unsigned short* tp = &t_lds[p * L_ * C_];
    size_t obase = (size_t)b * C_ * HW_ + (size_t)h * W_ + (size_t)(w0 + p);
#pragma unroll 4
    for (int j = 0; j < 16; ++j) {
        int c = cc * 16 + j;
        float v = 0.0f;
#pragma unroll
        for (int m = 0; m < L_; ++m) v += wgt[m] * bf2f(tp[m * C_ + c]);
        out[obase + (size_t)c * HW_] = v;
    }
}

extern "C" void kernel_launch(void* const* d_in, const int* in_sizes, int n_in,
                              void* d_out, int out_size, void* d_ws, size_t ws_size,
                              hipStream_t stream) {
    (void)in_sizes; (void)n_in; (void)out_size; (void)ws_size;
    const float* x   = (const float*)d_in[0];
    const float* rm  = (const float*)d_in[1];
    // d_in[2] = record_len (uniform L, unused)
    const float* T44 = (const float*)d_in[3];
    float* out = (float*)d_out;
    float* wsf = (float*)d_ws;

    rate_partial_kernel<<<B_, 256, 0, stream>>>(rm, wsf);
    rate_final_kernel<<<1, 1, 0, stream>>>(wsf, out);

    int blocks = B_ * H_ * (W_ / TILE);                // 4096
    fuse_kernel<<<blocks, THREADS, 0, stream>>>(x, rm, T44, out);
}
